// OurModel_20048907338198
// MI455X (gfx1250) — matrix operand order repaired and checked by reading, not compiled, not run on hardware
//
#include <hip/hip_runtime.h>
#include <hip/hip_bf16.h>

// ---------------------------------------------------------------- problem dims
#define SS 512
#define BB 256
#define II 227
#define HH 100
#define TT 19
#define H2 200   // 2*H

#define NTH 7    // H padded to 112 -> 7 n-tiles of 16
#define NTT 2    // T padded to 32 -> 2 n-tiles of 16

typedef __attribute__((ext_vector_type(16))) __bf16 v16bf;
typedef __attribute__((ext_vector_type(8)))  __bf16 v8bf;
typedef __attribute__((ext_vector_type(8)))  float  v8f;
typedef __attribute__((ext_vector_type(4)))  float  v4f;

// workspace byte offsets
#define OFF_WIH   0u            // 2*7*8*32*16 bf16 = 114688 B
#define OFF_WHH   114688u       // 2*7*4*32*16 bf16 = 57344 B
#define OFF_WTAG  172032u       // 2*7*32*16 bf16 = 14336 B
#define OFF_XPROJ 262144u       // 2*S*B*H f32 = 104857600 B
#define OFF_HS    105119744u    // S*B*2H f32 = 104857600 B
#define OFF_EM    209977344u    // S*B*T f32 = 9961472 B

__device__ __forceinline__ unsigned short f2bfu(float f) {
  unsigned u = __builtin_bit_cast(unsigned, f);
  u += 0x7FFFu + ((u >> 16) & 1u);
  return (unsigned short)(u >> 16);
}
__device__ __forceinline__ __bf16 f2bf(float f) {
  unsigned short h = f2bfu(f);
  return __builtin_bit_cast(__bf16, h);
}
__device__ __forceinline__ v8f wmma_bf16(v16bf a, v16bf b, v8f c) {
  return __builtin_amdgcn_wmma_f32_16x16x32_bf16(false, a, false, b, (short)0, c,
                                                 false, false);
}

// Issue one 16x100 f32 tile (6400 B, contiguous) global -> LDS as 13 async
// b128 transfers per lane group (index clamped so EXEC stays full and exactly
// 13 ASYNCcnt increments occur; duplicate transfers are benign).
__device__ __forceinline__ void issue_xtile(
    const float* gbase, __attribute__((address_space(3))) const float* lbase,
    int lane) {
  unsigned long long g0 = (unsigned long long)gbase;
  unsigned l0 = (unsigned)(size_t)lbase;
#pragma unroll
  for (int i = 0; i < 13; ++i) {
    int idx = lane + i * 32;
    if (idx > 399) idx = 399;  // 400 b128 transfers cover 6400 B
    unsigned long long ga = g0 + (unsigned long long)(idx * 16);
    unsigned la = l0 + (unsigned)(idx * 16);
    asm volatile("global_load_async_to_lds_b128 %0, %1, off"
                 :
                 : "v"(la), "v"(ga)
                 : "memory");
  }
}

// ---------------------------------------------------------------- weight packing
// B-fragment layout: element e of lane L holds B[k][n] with
// n = ntile*16 + (L&15), k = ktile*32 + (L>>4)*8 + (e&7) + (e>>3)*16.
__global__ void pack_kernel(const float* __restrict__ wihf,
                            const float* __restrict__ wihb,
                            const float* __restrict__ whhf,
                            const float* __restrict__ whhb,
                            const float* __restrict__ wtag,
                            unsigned short* __restrict__ pack) {
  int idx = blockIdx.x * blockDim.x + threadIdx.x;
  if (idx >= 93184) return;
  float v = 0.f;
  if (idx < 57344) {                       // W_ih: [dir][nt(7)][kt(8)][lane][16]
    int e = idx & 15; int q = idx >> 4;
    int lane = q & 31; q >>= 5;
    int kt = q & 7;    q >>= 3;
    int nt = q % 7;    int dir = q / 7;
    int n = nt * 16 + (lane & 15);
    int k = kt * 32 + (lane >> 4) * 8 + (e & 7) + (e >> 3) * 16;
    const float* w = dir ? wihb : wihf;
    if (n < HH && k < II) v = w[n * II + k];
    pack[idx] = f2bfu(v);
  } else if (idx < 86016) {                // W_hh: [dir][nt(7)][kt(4)][lane][16]
    int i2 = idx - 57344;
    int e = i2 & 15; int q = i2 >> 4;
    int lane = q & 31; q >>= 5;
    int kt = q & 3;    q >>= 2;
    int nt = q % 7;    int dir = q / 7;
    int n = nt * 16 + (lane & 15);
    int k = kt * 32 + (lane >> 4) * 8 + (e & 7) + (e >> 3) * 16;
    const float* w = dir ? whhb : whhf;
    if (n < HH && k < HH) v = w[n * HH + k];
    pack[idx] = f2bfu(v);
  } else {                                 // W_tag: [nt(2)][kt(7)][lane][16]
    int i3 = idx - 86016;
    int e = i3 & 15; int q = i3 >> 4;
    int lane = q & 31; q >>= 5;
    int kt = q % 7;    int nt = q / 7;
    int n = nt * 16 + (lane & 15);
    int k = kt * 32 + (lane >> 4) * 8 + (e & 7) + (e >> 3) * 16;
    if (n < TT && k < H2) v = wtag[n * H2 + k];
    pack[idx] = f2bfu(v);
  }
}

// ---------------------------------------------------------------- input projection
__global__ __launch_bounds__(256) void proj_kernel(
    const float* __restrict__ x, const unsigned short* __restrict__ packWih,
    const float* __restrict__ bihf, const float* __restrict__ bihb,
    float* __restrict__ xproj) {
  const int dir = blockIdx.y;
  const int wave = threadIdx.x >> 5, lane = threadIdx.x & 31;
  const int mrow = lane & 15, hiq = lane >> 4;
  const int r0 = (blockIdx.x * 8 + wave) * 16;   // 131072 rows total, exact
  const __bf16 bz = __builtin_bit_cast(__bf16, (unsigned short)0);
  const v16bf* bw = (const v16bf*)packWih;

  v8f acc[NTH];
  const v8f z = {0.f, 0.f, 0.f, 0.f, 0.f, 0.f, 0.f, 0.f};
#pragma unroll
  for (int nt = 0; nt < NTH; ++nt) acc[nt] = z;

  const size_t rbase = (size_t)(r0 + mrow) * II;
#pragma unroll
  for (int kt = 0; kt < 8; ++kt) {
    const int kA = kt * 32 + (hiq ? 8 : 0);
    v16bf a;
#pragma unroll
    for (int e = 0; e < 8; ++e) {
      int k = kA + e;
      a[e] = (k < II) ? f2bf(x[rbase + k]) : bz;
    }
#pragma unroll
    for (int e = 0; e < 8; ++e) {
      int k = kA + 16 + e;
      a[8 + e] = (k < II) ? f2bf(x[rbase + k]) : bz;
    }
#pragma unroll
    for (int nt = 0; nt < NTH; ++nt) {
      v16bf b = bw[((dir * NTH + nt) * 8 + kt) * 32 + lane];
      acc[nt] = wmma_bf16(a, b, acc[nt]);
    }
  }

  const float* bih = dir ? bihb : bihf;
  float* xp = xproj + (size_t)dir * SS * BB * HH;
#pragma unroll
  for (int nt = 0; nt < NTH; ++nt) {
    int n = nt * 16 + mrow;
    if (n < HH) {
      float bv = bih[n];
#pragma unroll
      for (int j = 0; j < 8; ++j) {
        int m = j + hiq * 8;
        xp[(size_t)(r0 + m) * HH + n] = acc[nt][j] + bv;
      }
    }
  }
}

// ---------------------------------------------------------------- recurrent scan
// One wave32 per (dir, 16-batch chunk). Hidden tile and w_hh fragments live in
// LDS; per-step xproj tile is double-buffered in LDS via async b128 loads that
// overlap the 28 WMMAs of the current step.
__global__ __launch_bounds__(32) void rnn_kernel(
    const float* __restrict__ xproj, const unsigned short* __restrict__ packWhh,
    const float* __restrict__ bhhf, const float* __restrict__ bhhb,
    float* __restrict__ hs) {
  const int dir = blockIdx.y;
  const int b0 = blockIdx.x * 16;
  const int lane = threadIdx.x & 31;
  const int mrow = lane & 15, hiq = lane >> 4;
  const __bf16 bz = __builtin_bit_cast(__bf16, (unsigned short)0);

  __shared__ __attribute__((aligned(32))) __bf16 hl[16 * 128];
  __shared__ __attribute__((aligned(32))) __bf16 wsh[28 * 32 * 16];
  __shared__ __attribute__((aligned(16))) float xlds[2 * 16 * HH];

  for (int i = lane; i < 16 * 128; i += 32) hl[i] = bz;
  { // stage w_hh fragments for this direction into LDS
    const v16bf* wg = (const v16bf*)packWhh;
    v16bf* wl = (v16bf*)wsh;
#pragma unroll
    for (int f = 0; f < 28; ++f) wl[f * 32 + lane] = wg[(dir * 28 + f) * 32 + lane];
  }

  const float* bh = dir ? bhhb : bhhf;
  float bias[NTH];
#pragma unroll
  for (int nt = 0; nt < NTH; ++nt) {
    int n = nt * 16 + mrow;
    bias[nt] = (n < HH) ? bh[n] : 0.f;
  }
  const float* xp = xproj + (size_t)dir * SS * BB * HH;
  const v16bf* wl = (const v16bf*)wsh;
  __attribute__((address_space(3))) const float* xl3 =
      (__attribute__((address_space(3))) const float*)xlds;

  // prologue: fetch tile for t = 0 into buffer 0
  {
    const int s0 = dir ? (SS - 1) : 0;
    issue_xtile(xp + (size_t)(s0 * BB + b0) * HH, xl3, lane);
  }

  for (int t = 0; t < SS; ++t) {
    const int s = dir ? (SS - 1 - t) : t;
    // kick off next step's tile into the other buffer (clamped at the end)
    const int tn = (t + 1 < SS) ? (t + 1) : t;
    const int sn = dir ? (SS - 1 - tn) : tn;
    issue_xtile(xp + (size_t)(sn * BB + b0) * HH, xl3 + ((t + 1) & 1) * 16 * HH,
                lane);
    // wait until only the 13 just-issued transfers remain -> buffer t ready
    asm volatile("s_wait_asynccnt 0xd" ::: "memory");

    // A fragments (h from previous step) from LDS
    v16bf afr[4];
#pragma unroll
    for (int kt = 0; kt < 4; ++kt) {
      const int kA = kt * 32 + (hiq ? 8 : 0);
      v8bf lo = *(const v8bf*)(&hl[mrow * 128 + kA]);
      v8bf hi = *(const v8bf*)(&hl[mrow * 128 + kA + 16]);
      afr[kt] = __builtin_shufflevector(lo, hi, 0, 1, 2, 3, 4, 5, 6, 7, 8, 9,
                                        10, 11, 12, 13, 14, 15);
    }
    const float* xt = &xlds[(t & 1) * 16 * HH];
#pragma unroll
    for (int nt = 0; nt < NTH; ++nt) {
      v8f c = {0.f, 0.f, 0.f, 0.f, 0.f, 0.f, 0.f, 0.f};
#pragma unroll
      for (int kt = 0; kt < 4; ++kt)
        c = wmma_bf16(afr[kt], wl[(nt * 4 + kt) * 32 + lane], c);
      const int n = nt * 16 + mrow;
      const bool nv = (n < HH);
      const float bv = bias[nt];
#pragma unroll
      for (int j = 0; j < 8; ++j) {
        const int m = j + hiq * 8;
        const size_t row = (size_t)(s * BB + b0 + m);
        float xv = nv ? xt[m * HH + n] : 0.f;
        float d = tanhf(c[j] + xv + bv);
        if (nv) {
          hs[row * H2 + dir * HH + n] = d;       // concat(hf, hb)
          hl[m * 128 + n] = f2bf(d);             // next step's A operand
        }
      }
    }
  }
}

// ---------------------------------------------------------------- emissions
__global__ __launch_bounds__(256) void emis_kernel(
    const float* __restrict__ hs, const unsigned short* __restrict__ packWtag,
    const float* __restrict__ btag, float* __restrict__ em) {
  const int wave = threadIdx.x >> 5, lane = threadIdx.x & 31;
  const int mrow = lane & 15, hiq = lane >> 4;
  const int r0 = (blockIdx.x * 8 + wave) * 16;
  const __bf16 bz = __builtin_bit_cast(__bf16, (unsigned short)0);
  const v16bf* bw = (const v16bf*)packWtag;

  v8f acc[NTT];
  const v8f z = {0.f, 0.f, 0.f, 0.f, 0.f, 0.f, 0.f, 0.f};
#pragma unroll
  for (int nt = 0; nt < NTT; ++nt) acc[nt] = z;

  const float* hsrow = hs + (size_t)(r0 + mrow) * H2;  // 800B-aligned rows
#pragma unroll
  for (int kt = 0; kt < 7; ++kt) {
    const int kA = kt * 32 + (hiq ? 8 : 0);
    v16bf a;
#pragma unroll
    for (int half = 0; half < 2; ++half) {
      const int k0 = kA + half * 16;          // 32B-aligned run start
      if (k0 + 8 <= H2) {                     // fully in-bounds: 2x b128
        v4f u0 = *(const v4f*)(hsrow + k0);
        v4f u1 = *(const v4f*)(hsrow + k0 + 4);
#pragma unroll
        for (int e = 0; e < 4; ++e) a[half * 8 + e] = f2bf(u0[e]);
#pragma unroll
        for (int e = 0; e < 4; ++e) a[half * 8 + 4 + e] = f2bf(u1[e]);
      } else {                                // tail: guarded elementwise
#pragma unroll
        for (int e = 0; e < 8; ++e) {
          int k = k0 + e;
          a[half * 8 + e] = (k < H2) ? f2bf(hsrow[k]) : bz;
        }
      }
    }
#pragma unroll
    for (int nt = 0; nt < NTT; ++nt) {
      v16bf b = bw[(nt * 7 + kt) * 32 + lane];
      acc[nt] = wmma_bf16(a, b, acc[nt]);
    }
  }
#pragma unroll
  for (int nt = 0; nt < NTT; ++nt) {
    int n = nt * 16 + mrow;
    if (n < TT) {
      float bv = btag[n];
#pragma unroll
      for (int j = 0; j < 8; ++j) {
        int m = j + hiq * 8;
        em[(size_t)(r0 + m) * TT + n] = acc[nt][j] + bv;
      }
    }
  }
}

// ---------------------------------------------------------------- CRF
__global__ __launch_bounds__(256) void crf_kernel(
    const float* __restrict__ em, const int* __restrict__ tags,
    const float* __restrict__ startt, const float* __restrict__ endt,
    const float* __restrict__ trans, float* __restrict__ out) {
  __shared__ float trT[TT * TT];  // trT[t'][t] = trans[t][t']
  __shared__ float red[256];
  const int tid = threadIdx.x;
  const int b = tid;
  for (int i = tid; i < TT * TT; i += 256) {
    int tp = i / TT, t = i % TT;
    trT[i] = trans[t * TT + tp];
  }
  __syncthreads();

  int tag = tags[b];  // s = 0
  float numer = startt[tag] + em[(size_t)b * TT + tag];
  int prev = tag;

  float score[TT];
#pragma unroll
  for (int t = 0; t < TT; ++t) score[t] = startt[t] + em[(size_t)b * TT + t];

  for (int s = 1; s < SS; ++s) {
    const size_t base = (size_t)(s * BB + b) * TT;
    tag = tags[s * BB + b];
    numer += em[base + tag] + trans[prev * TT + tag];
    prev = tag;

    float emr[TT];
#pragma unroll
    for (int t = 0; t < TT; ++t) emr[t] = em[base + t];

    float nxt[TT];
#pragma unroll
    for (int tp = 0; tp < TT; ++tp) {
      float mx = -3.0e38f;
#pragma unroll
      for (int t = 0; t < TT; ++t)
        mx = fmaxf(mx, score[t] + trT[tp * TT + t]);
      float sm = 0.f;
#pragma unroll
      for (int t = 0; t < TT; ++t)
        sm += __expf(score[t] + trT[tp * TT + t] - mx);
      nxt[tp] = emr[tp] + mx + __logf(sm);
    }
#pragma unroll
    for (int t = 0; t < TT; ++t) score[t] = nxt[t];
  }
  numer += endt[prev];

  float mx = -3.0e38f;
#pragma unroll
  for (int t = 0; t < TT; ++t) mx = fmaxf(mx, score[t] + endt[t]);
  float sm = 0.f;
#pragma unroll
  for (int t = 0; t < TT; ++t) sm += __expf(score[t] + endt[t] - mx);
  float denom = mx + __logf(sm);

  red[tid] = numer - denom;
  __syncthreads();
#pragma unroll
  for (int off = 128; off > 0; off >>= 1) {
    if (tid < off) red[tid] += red[tid + off];
    __syncthreads();
  }
  if (tid == 0) out[0] = red[0];
}

// ---------------------------------------------------------------- launcher
extern "C" void kernel_launch(void* const* d_in, const int* in_sizes, int n_in,
                              void* d_out, int out_size, void* d_ws,
                              size_t ws_size, hipStream_t stream) {
  const float* x      = (const float*)d_in[0];
  const int*   tags   = (const int*)d_in[1];
  const float* w_ih_f = (const float*)d_in[2];
  const float* w_hh_f = (const float*)d_in[3];
  const float* b_ih_f = (const float*)d_in[4];
  const float* b_hh_f = (const float*)d_in[5];
  const float* w_ih_b = (const float*)d_in[6];
  const float* w_hh_b = (const float*)d_in[7];
  const float* b_ih_b = (const float*)d_in[8];
  const float* b_hh_b = (const float*)d_in[9];
  const float* W_tag  = (const float*)d_in[10];
  const float* b_tag  = (const float*)d_in[11];
  const float* start_t = (const float*)d_in[12];
  const float* end_t   = (const float*)d_in[13];
  const float* trans   = (const float*)d_in[14];

  char* ws = (char*)d_ws;
  unsigned short* packWih  = (unsigned short*)(ws + OFF_WIH);
  unsigned short* packWhh  = (unsigned short*)(ws + OFF_WHH);
  unsigned short* packWtag = (unsigned short*)(ws + OFF_WTAG);
  float* xproj = (float*)(ws + OFF_XPROJ);
  float* hs    = (float*)(ws + OFF_HS);
  float* em    = (float*)(ws + OFF_EM);

  pack_kernel<<<(93184 + 255) / 256, 256, 0, stream>>>(w_ih_f, w_ih_b, w_hh_f,
                                                       w_hh_b, W_tag, packWih);
  proj_kernel<<<dim3(1024, 2), 256, 0, stream>>>(x, packWih, b_ih_f, b_ih_b,
                                                 xproj);
  rnn_kernel<<<dim3(16, 2), 32, 0, stream>>>(xproj, packWhh, b_hh_f, b_hh_b, hs);
  emis_kernel<<<1024, 256, 0, stream>>>(hs, packWtag, b_tag, em);
  crf_kernel<<<1, 256, 0, stream>>>(em, tags, start_t, end_t, trans,
                                    (float*)d_out);
}
